// MultiheadAttention_78417512891235
// MI455X (gfx1250) — compile-verified, matrix operand order checked by reading, not run on hardware
//
#include <hip/hip_runtime.h>
#include <hip/hip_bf16.h>
#include <math.h>

typedef __attribute__((ext_vector_type(16))) __bf16 v16bf;
typedef __attribute__((ext_vector_type(8)))  __bf16 bf16x8;
typedef __attribute__((ext_vector_type(8)))  float  v8f;
typedef __attribute__((ext_vector_type(4)))  float  f32x4;
typedef __attribute__((ext_vector_type(4)))  int    v4i;

#define D_MODEL   1024
#define NUM_HEADS 16
#define HEAD_DIM  64
#define BATCH     4
#define SEQ       2048

// Async global->LDS DMA path (gfx1250): probe-guarded. The toolchain's
// signature (from the round-2 diagnostic) is:
//   void __builtin_amdgcn_global_load_async_to_lds_b128(
//       int4 addrspace(1)* src, int4 addrspace(3)* dst, int off, int cpol)
#if defined(__has_builtin)
#  if __has_builtin(__builtin_amdgcn_global_load_async_to_lds_b128) && \
      __has_builtin(__builtin_amdgcn_s_wait_asynccnt)
#    define HAVE_ASYNC_LDS 1
#  endif
#endif

typedef v4i __attribute__((address_space(1)))* gptr_v4i;
typedef v4i __attribute__((address_space(3)))* lptr_v4i;

// ---------------------------------------------------------------------------
// WMMA helper: D = A(16x32 bf16) x B(32x16 bf16) + C(16x16 f32)
// ---------------------------------------------------------------------------
__device__ __forceinline__ v8f wmma_bf16(v16bf a, v16bf b, v8f c) {
    return __builtin_amdgcn_wmma_f32_16x16x32_bf16(
        /*neg_a=*/false, a, /*neg_b=*/false, b,
        /*c_mod=*/(short)0, c, /*reuse_a=*/false, /*reuse_b=*/false);
}

// A-fragment loader for a 16x32 bf16 tile slice starting at column k0.
// Per ISA layout: lane half 'hi' supplies K chunks [k0+hi*8, +8) and
// [k0+16+hi*8, +8)  -> two 16-byte loads per lane.
__device__ __forceinline__ v16bf load_frag(const __bf16* base, int row,
                                           int stride, int k0, int hi) {
    const bf16x8 c0 = *(const bf16x8*)(base + (size_t)row * stride + k0 + hi * 8);
    const bf16x8 c1 = *(const bf16x8*)(base + (size_t)row * stride + k0 + 16 + hi * 8);
    v16bf r;
#pragma unroll
    for (int i = 0; i < 8; ++i) { r[i] = c0[i]; r[i + 8] = c1[i]; }
    return r;
}

// ---------------------------------------------------------------------------
// Kernel 1: fused QKV projection.  C[M=8192, N=1024] = X @ W^T, written as
// bf16 in head-major (B, H, S, 64) layout.  grid.z selects q/k/v.
// BM=128, BN=64, BK=32; 256 threads = 8 waves, each wave -> 32x32 tile.
// ---------------------------------------------------------------------------
__global__ __launch_bounds__(256)
void qkv_proj_kernel(const float* __restrict__ q, const float* __restrict__ k,
                     const float* __restrict__ v, const float* __restrict__ Wq,
                     const float* __restrict__ Wk, const float* __restrict__ Wv,
                     __bf16* __restrict__ Qh, __bf16* __restrict__ Kh,
                     __bf16* __restrict__ Vh) {
    const int which = blockIdx.z;
    const float* X = (which == 0) ? q : (which == 1) ? k : v;
    const float* W = (which == 0) ? Wq : (which == 1) ? Wk : Wv;
    __bf16*      O = (which == 0) ? Qh : (which == 1) ? Kh : Vh;

    constexpr int BM = 128, BN = 64, BK = 32;
    constexpr int LDA = 40, LDB = 40;   // halves; 80B rows keep 16B chunk align
    __shared__ __bf16 sA[BM * LDA];
    __shared__ __bf16 sB[BN * LDB];

    const int tid  = threadIdx.x;
    const int lane = tid & 31;
    const int wave = tid >> 5;       // 0..7
    const int wm   = wave >> 1;      // 0..3 : 32-row band
    const int wn   = wave & 1;       // 0..1 : 32-col band
    const int hi   = (lane >> 4) & 1;
    const int lr   = lane & 15;

    const int gm0 = blockIdx.y * BM;
    const int gn0 = blockIdx.x * BN;

    v8f acc[2][2];
#pragma unroll
    for (int i = 0; i < 2; ++i)
#pragma unroll
        for (int j = 0; j < 2; ++j) acc[i][j] = {};

    for (int k0 = 0; k0 < D_MODEL; k0 += BK) {
        __syncthreads();
        // Stage A: 128x32 fp32 -> bf16.  1024 float4 chunks / 256 threads.
#pragma unroll
        for (int i = 0; i < 4; ++i) {
            int c = tid * 4 + i;
            int row = c >> 3, off = (c & 7) * 4;
            f32x4 d = *(const f32x4*)(X + (size_t)(gm0 + row) * D_MODEL + k0 + off);
            sA[row * LDA + off + 0] = (__bf16)d.x;
            sA[row * LDA + off + 1] = (__bf16)d.y;
            sA[row * LDA + off + 2] = (__bf16)d.z;
            sA[row * LDA + off + 3] = (__bf16)d.w;
        }
        // Stage B: 64x32 rows of W (row e of W = column e of W^T).
#pragma unroll
        for (int i = 0; i < 2; ++i) {
            int c = tid * 2 + i;
            int row = c >> 3, off = (c & 7) * 4;
            f32x4 d = *(const f32x4*)(W + (size_t)(gn0 + row) * D_MODEL + k0 + off);
            sB[row * LDB + off + 0] = (__bf16)d.x;
            sB[row * LDB + off + 1] = (__bf16)d.y;
            sB[row * LDB + off + 2] = (__bf16)d.z;
            sB[row * LDB + off + 3] = (__bf16)d.w;
        }
        __syncthreads();

        // Prefetch next K slab into L2 while we compute.
        if (k0 + BK < D_MODEL) {
            __builtin_prefetch(X + (size_t)(gm0 + (tid >> 1)) * D_MODEL + k0 + BK +
                                   (tid & 1) * 16, 0, 1);
        }

        v16bf a0 = load_frag(sA, wm * 32 + 0  + lr, LDA, 0, hi);
        v16bf a1 = load_frag(sA, wm * 32 + 16 + lr, LDA, 0, hi);
        v16bf b0 = load_frag(sB, wn * 32 + 0  + lr, LDB, 0, hi);
        v16bf b1 = load_frag(sB, wn * 32 + 16 + lr, LDB, 0, hi);
        acc[0][0] = wmma_bf16(a0, b0, acc[0][0]);
        acc[0][1] = wmma_bf16(a0, b1, acc[0][1]);
        acc[1][0] = wmma_bf16(a1, b0, acc[1][0]);
        acc[1][1] = wmma_bf16(a1, b1, acc[1][1]);
    }

    // Epilogue: write bf16 head-major (B, H, S, 64).
#pragma unroll
    for (int i = 0; i < 2; ++i)
#pragma unroll
        for (int j = 0; j < 2; ++j)
#pragma unroll
            for (int r = 0; r < 8; ++r) {
                int m = gm0 + wm * 32 + i * 16 + r + hi * 8;   // row in (B*S)
                int n = gn0 + wn * 32 + j * 16 + lr;           // col in D_MODEL
                int bb = m >> 11, s = m & (SEQ - 1);
                int h = n >> 6, dh = n & (HEAD_DIM - 1);
                O[(((size_t)(bb * NUM_HEADS + h)) * SEQ + s) * HEAD_DIM + dh] =
                    (__bf16)acc[i][j][r];
            }
}

// ---------------------------------------------------------------------------
// Kernel 2: flash attention over bf16 head-major Q/K/V.
// grid = (SEQ/64, B*H); 128 threads = 4 waves, each wave owns 16 query rows.
// ---------------------------------------------------------------------------
__global__ __launch_bounds__(128)
void flash_attn_kernel(const __bf16* __restrict__ Qh,
                       const __bf16* __restrict__ Kh,
                       const __bf16* __restrict__ Vh,
                       float* __restrict__ out) {
    constexpr int LDK = 72, LDV = 72, LDP = 72;   // halves; 144B rows, 16B ok
    __shared__ __bf16 sK [64 * LDK];   // K block, row-major (key x dh)
    __shared__ __bf16 sVt[64 * LDV];   // V block, transposed (dh x key)
    __shared__ __bf16 sP [64 * LDP];   // per-wave P staging (4 x 16 rows)

    const int tid  = threadIdx.x;
    const int lane = tid & 31;
    const int wave = tid >> 5;         // 0..3
    const int hi   = (lane >> 4) & 1;
    const int lr   = lane & 15;
    const int bh   = blockIdx.y;                 // b*16 + h
    const int q0   = blockIdx.x * 64 + wave * 16;

    const __bf16* Qb = Qh + (size_t)bh * SEQ * HEAD_DIM;
    const __bf16* Kb = Kh + (size_t)bh * SEQ * HEAD_DIM;
    const __bf16* Vb = Vh + (size_t)bh * SEQ * HEAD_DIM;

    // Q fragments for this wave's 16 rows (K-dim = head dim, 2 x 32).
    v16bf qa0 = load_frag(Qb, q0 + lr, HEAD_DIM, 0,  hi);
    v16bf qa1 = load_frag(Qb, q0 + lr, HEAD_DIM, 32, hi);

    v8f acc[4];
#pragma unroll
    for (int n = 0; n < 4; ++n) acc[n] = {};
    float mrow[8], lrow[8];
#pragma unroll
    for (int r = 0; r < 8; ++r) { mrow[r] = -INFINITY; lrow[r] = 0.f; }

    const float scale = 0.125f;   // 1/sqrt(64)

    for (int j0 = 0; j0 < SEQ; j0 += 64) {
        __syncthreads();
        // Stage K block row-major: 512 x 16B chunks / 128 threads.
#ifdef HAVE_ASYNC_LDS
        // Async DMA path: GLOBAL_LOAD_ASYNC_TO_LDS_B128, tracked by ASYNCcnt.
#pragma unroll
        for (int i = 0; i < 4; ++i) {
            int c = tid * 4 + i;
            int row = c >> 3, off = (c & 7) * 8;
            __builtin_amdgcn_global_load_async_to_lds_b128(
                (gptr_v4i)(Kb + (size_t)(j0 + row) * HEAD_DIM + off),
                (lptr_v4i)(sK + row * LDK + off),
                0, 0);
        }
#else
#pragma unroll
        for (int i = 0; i < 4; ++i) {
            int c = tid * 4 + i;
            int row = c >> 3, off = (c & 7) * 8;
            *(bf16x8*)(sK + row * LDK + off) =
                *(const bf16x8*)(Kb + (size_t)(j0 + row) * HEAD_DIM + off);
        }
#endif
        // Stage V block transposed: vector global load, scalar LDS scatter
        // (overlaps with the async K DMA above when available).
#pragma unroll
        for (int i = 0; i < 4; ++i) {
            int c = tid * 4 + i;
            int row = c >> 3, off = (c & 7) * 8;
            bf16x8 d = *(const bf16x8*)(Vb + (size_t)(j0 + row) * HEAD_DIM + off);
#pragma unroll
            for (int e = 0; e < 8; ++e) sVt[(off + e) * LDV + row] = d[e];
        }
#ifdef HAVE_ASYNC_LDS
        __builtin_amdgcn_s_wait_asynccnt(0);
#endif
        __syncthreads();

        if (j0 + 64 < SEQ)
            __builtin_prefetch(Kb + (size_t)(j0 + 64 + (tid >> 1)) * HEAD_DIM +
                                   (tid & 1) * 32, 0, 1);

        // Scores: S = Q (16x64) @ K^T (64x16) per 16-wide key tile.
        v8f sc[4];
#pragma unroll
        for (int n = 0; n < 4; ++n) {
            v16bf kb0 = load_frag(sK, n * 16 + lr, LDK, 0,  hi);
            v16bf kb1 = load_frag(sK, n * 16 + lr, LDK, 32, hi);
            v8f z = {};
            z = wmma_bf16(qa0, kb0, z);
            z = wmma_bf16(qa1, kb1, z);
            sc[n] = z;
        }

        // Online softmax. C-layout: lanes 0-15 hold rows r, 16-31 rows r+8;
        // row values live across the 16 lanes of each accumulator register.
#pragma unroll
        for (int r = 0; r < 8; ++r) {
#pragma unroll
            for (int n = 0; n < 4; ++n) sc[n][r] *= scale;
            float rm = fmaxf(fmaxf(sc[0][r], sc[1][r]), fmaxf(sc[2][r], sc[3][r]));
            rm = fmaxf(rm, __shfl_xor(rm, 1, 32));
            rm = fmaxf(rm, __shfl_xor(rm, 2, 32));
            rm = fmaxf(rm, __shfl_xor(rm, 4, 32));
            rm = fmaxf(rm, __shfl_xor(rm, 8, 32));
            float mnew  = fmaxf(mrow[r], rm);
            float alpha = __expf(mrow[r] - mnew);
            float rs = 0.f;
#pragma unroll
            for (int n = 0; n < 4; ++n) {
                float p = __expf(sc[n][r] - mnew);
                sc[n][r] = p;
                rs += p;
            }
            rs += __shfl_xor(rs, 1, 32);
            rs += __shfl_xor(rs, 2, 32);
            rs += __shfl_xor(rs, 4, 32);
            rs += __shfl_xor(rs, 8, 32);
            lrow[r] = lrow[r] * alpha + rs;
            mrow[r] = mnew;
#pragma unroll
            for (int n = 0; n < 4; ++n) acc[n][r] *= alpha;
        }

        // Re-shape P: C-layout -> LDS (wave-private) -> A-fragment layout.
#pragma unroll
        for (int n = 0; n < 4; ++n)
#pragma unroll
            for (int r = 0; r < 8; ++r)
                sP[(wave * 16 + r + hi * 8) * LDP + n * 16 + lr] = (__bf16)sc[n][r];

        v16bf pa0 = load_frag(sP, wave * 16 + lr, LDP, 0,  hi);
        v16bf pa1 = load_frag(sP, wave * 16 + lr, LDP, 32, hi);

        // O += P (16x64) @ V (64x64); V supplied as V^T rows from sVt.
#pragma unroll
        for (int n = 0; n < 4; ++n) {
            v16bf vb0 = load_frag(sVt, n * 16 + lr, LDV, 0,  hi);
            v16bf vb1 = load_frag(sVt, n * 16 + lr, LDV, 32, hi);
            acc[n] = wmma_bf16(pa0, vb0, acc[n]);
            acc[n] = wmma_bf16(pa1, vb1, acc[n]);
        }
    }

    // Epilogue: out is the raw (B,H,S,Dh) flattening (reference's .view).
#pragma unroll
    for (int r = 0; r < 8; ++r) {
        float inv = 1.0f / lrow[r];
        int s = q0 + r + hi * 8;
#pragma unroll
        for (int n = 0; n < 4; ++n)
            out[((size_t)bh * SEQ + s) * HEAD_DIM + n * 16 + lr] = acc[n][r] * inv;
    }
}

// ---------------------------------------------------------------------------
// Host launcher
// ---------------------------------------------------------------------------
extern "C" void kernel_launch(void* const* d_in, const int* in_sizes, int n_in,
                              void* d_out, int out_size, void* d_ws, size_t ws_size,
                              hipStream_t stream) {
    const float* q  = (const float*)d_in[0];
    const float* k  = (const float*)d_in[1];
    const float* v  = (const float*)d_in[2];
    const float* Wq = (const float*)d_in[3];
    const float* Wk = (const float*)d_in[4];
    const float* Wv = (const float*)d_in[5];

    const size_t HSZ = (size_t)BATCH * NUM_HEADS * SEQ * HEAD_DIM;  // 8388608
    __bf16* Qh = (__bf16*)d_ws;
    __bf16* Kh = Qh + HSZ;
    __bf16* Vh = Kh + HSZ;

    // 1) Fused QKV projection into head-major bf16 workspace.
    dim3 gProj(D_MODEL / 64, (BATCH * SEQ) / 128, 3);
    qkv_proj_kernel<<<gProj, 256, 0, stream>>>(q, k, v, Wq, Wk, Wv, Qh, Kh, Vh);

    // 2) Flash attention per (64-query-block, head).
    dim3 gAttn(SEQ / 64, BATCH * NUM_HEADS);
    flash_attn_kernel<<<gAttn, 128, 0, stream>>>(Qh, Kh, Vh, (float*)d_out);
}
